// GATCNN_type1_62113817035132
// MI455X (gfx1250) — matrix operand order repaired and checked by reading, not compiled, not run on hardware
//
#include <hip/hip_runtime.h>
#include <stdint.h>

// ---------------------------------------------------------------------------
// GAT-CNN block for MI455X (gfx1250, wave32).
//   1) convert_x    : x f32 -> bf16 (RNE), layout preserved [B,256,HW]
//   2) pack_weights : Wq|Wk|Wv -> bf16 [768,256], bq|bk|bv -> f32 [768]
//   3) gemm_qkv     : QKV = x * Wc^T via v_wmma_f32_16x16x32_bf16
//                     - A block-tile (128M x 256K) staged once in LDS [m][k],
//                       fragments read as 2x ds_load_b128 (A and B use
//                       contiguous K order per half: e0-7 -> K=half*16.., e8-15 ->
//                       K=half*16+8..)
//                     - W tile (64N x 256K) fetched by the Tensor Data Mover
//                       (tensor_load_to_lds, TENSORcnt) per ISA ch.8 D# layout
//   4) bn_stats     : per-channel mean/var over (B,H,W) -> folded scale/shift
//   5) attn         : 4-neighbor attention + softmax + ReLU + residual
// M_total = 512*361 = 184832 = 1444*128 (exact), N = 768 = 12*64, K = 256.
// Roofline: GEMMs are ~73 GFLOP; in bf16 WMMA they cost ~15-30us, so the
// pipeline is HBM-bound (~1.5-2 GB total traffic -> ~70-90us @ 23.3 TB/s).
// ---------------------------------------------------------------------------

typedef __attribute__((ext_vector_type(16))) __bf16 v16bf;
typedef __attribute__((ext_vector_type(8)))  float  v8f;
typedef __attribute__((ext_vector_type(4)))  unsigned int v4u;
typedef __attribute__((ext_vector_type(8)))  int  v8i;
typedef __attribute__((ext_vector_type(4)))  int  v4i;
typedef __attribute__((address_space(3))) unsigned short lds_us_t;

#define BATCH  512
#define IN_C   256
#define NHEAD  8
#define ATT_C  32
#define OUT_C  32
#define HH     19
#define WWD    19
#define HWS    (HH * WWD)          // 361
#define M_TOT  (BATCH * HWS)       // 184832
#define NQKV   768
#define KDIM   256
#define LDA    264                 // padded LDS row stride (elems) for A tile
#define QKV_B_STRIDE (NQKV * HWS)  // 277248
#define X_B_STRIDE   (IN_C * HWS)  // 92416

__device__ __forceinline__ unsigned short f32_to_bf16_rne(float f) {
    union { float f; unsigned int u; } v; v.f = f;
    unsigned int u = v.u;
    unsigned int r = u + 0x7FFFu + ((u >> 16) & 1u);      // round-to-nearest-even
    if ((u & 0x7F800000u) == 0x7F800000u) r = u;          // inf/nan: truncate
    return (unsigned short)(r >> 16);
}

// ---------------- 1) x -> bf16 (vectorized by 4) ---------------------------
__global__ __launch_bounds__(256) void convert_x(const float* __restrict__ x,
                                                 unsigned short* __restrict__ xb,
                                                 int n4) {
    int i = blockIdx.x * blockDim.x + threadIdx.x;
    if (i >= n4) return;
    const float4 v = ((const float4*)x)[i];
    ushort4 o;
    o.x = f32_to_bf16_rne(v.x);
    o.y = f32_to_bf16_rne(v.y);
    o.z = f32_to_bf16_rne(v.z);
    o.w = f32_to_bf16_rne(v.w);
    ((ushort4*)xb)[i] = o;
}

// ---------------- 2) pack Wq|Wk|Wv and biases ------------------------------
__global__ __launch_bounds__(256) void pack_weights(
    const float* __restrict__ Wq, const float* __restrict__ Wk,
    const float* __restrict__ Wv, const float* __restrict__ bq,
    const float* __restrict__ bk, const float* __restrict__ bv,
    unsigned short* __restrict__ wc, float* __restrict__ biasc) {
    int t = blockIdx.x * blockDim.x + threadIdx.x;   // 0 .. 768*256-1
    if (t >= NQKV * KDIM) return;
    int co = t / KDIM, ci = t - co * KDIM;
    float w = (co < 256) ? Wq[co * KDIM + ci]
            : (co < 512) ? Wk[(co - 256) * KDIM + ci]
                         : Wv[(co - 512) * KDIM + ci];
    wc[t] = f32_to_bf16_rne(w);
    if (t < NQKV) {
        biasc[t] = (t < 256) ? bq[t] : (t < 512) ? bk[t - 256] : bv[t - 512];
    }
}

// ---------------- 3) fused QKV GEMM via WMMA bf16 + TDM --------------------
// grid = (1444, 12), block = 256 threads = 8 waves.
// Wave w owns M rows [blk*128 + w*16, +16); block owns N cols [nblk*64, +64).
__global__ __launch_bounds__(256) void gemm_qkv(
    const unsigned short* __restrict__ xb,    // bf16 bits [B,256,HW] (A^T layout)
    const unsigned short* __restrict__ wc,    // bf16 bits [768,256]  (row-major)
    const float* __restrict__ biasc,          // [768]
    float* __restrict__ qkv) {                // f32 [B,768,HW]
    // LDS: A tile [128 m][264 k-padded] (67584 B) then W tile [64][256] (32768 B)
    __shared__ unsigned short smem[128 * LDA + 64 * KDIM];
    unsigned short* ldsA = smem;
    unsigned short* ldsW = smem + 128 * LDA;

    const int tid  = threadIdx.x;
    const int lane = tid & 31;
    const int wave = tid >> 5;
    const int half = lane >> 4;                     // K-half selector (A/B layouts)
    const int nlo  = lane & 15;
    const int Nbase = blockIdx.y * 64;

    // ---- issue TDM: W tile [64 rows x 256 bf16], row stride 256 -> ldsW ----
    if (wave == 0) {
        const unsigned lds_base =
            (unsigned)(unsigned long long)(lds_us_t*)smem;          // LDS byte offset
        const unsigned long long ga =
            (unsigned long long)(const void*)(wc + (size_t)Nbase * KDIM);
        v4u g0;
        g0.x = 1u;                                                  // count=1
        g0.y = lds_base + 128u * LDA * 2u;                          // lds_addr (bytes)
        g0.z = (unsigned)(ga & 0xFFFFFFFFull);                      // global_addr lo
        g0.w = (unsigned)((ga >> 32) & 0x01FFFFFFull) | (2u << 30); // addr hi | type=2
        v8i g1;
        g1[0] = 0x00010000;            // data_size=1 (2 bytes/elem)
        g1[1] = (int)(256u << 16);     // tensor_dim0 = 256 (bits 79:48, lo16)
        g1[2] = (int)(64u << 16);      // tensor_dim1 = 64  (bits 111:80, lo16)
        g1[3] = (int)(256u << 16);     // tile_dim0  = 256  (bits 127:112)
        g1[4] = 64;                    // tile_dim1  = 64   (bits 143:128)
        g1[5] = 256;                   // tensor_dim0_stride = 256 elems
        g1[6] = 0;
        g1[7] = 0;
        v4i g2 = {0, 0, 0, 0};         // 2D tensor: groups 2/3 unused
        v4i g3 = {0, 0, 0, 0};
        v8i g4 = {0, 0, 0, 0, 0, 0, 0, 0};  // 6-arg toolchain form: extra group
        __builtin_amdgcn_tensor_load_to_lds(g0, g1, g2, g3, g4, 0);
    }

    // ---- stage A tile transposed: global [k][m-contig] -> LDS [m][k] ----
    {
        const int mloc = tid & 127;
        const int mg   = blockIdx.x * 128 + mloc;
        const int bA   = mg / HWS;
        const int hwA  = mg - bA * HWS;
        const unsigned short* pA = xb + (size_t)bA * X_B_STRIDE + hwA;
        unsigned short* dst = ldsA + mloc * LDA;
        for (int k = tid >> 7; k < KDIM; k += 2) {  // coalesced u16 along m
            __builtin_prefetch(pA + (size_t)(k + 8) * HWS, 0, 1);
            dst[k] = pA[(size_t)k * HWS];
        }
    }
    if (wave == 0) __builtin_amdgcn_s_wait_tensorcnt(0);
    __syncthreads();

    // ---- K loop: fragments from LDS, 32 static WMMAs per wave ----
    const unsigned short* arow = ldsA + (wave * 16 + nlo) * LDA;  // A: M = lane%16
    v8f acc0 = {}, acc1 = {}, acc2 = {}, acc3 = {};
#pragma unroll
    for (int kb = 0; kb < KDIM; kb += 32) {
        // A fragment: e0-15 = K[kb+16h .. +15]
        union { v16bf v; uint4 q[2]; } af;
        af.q[0] = *(const uint4*)(arow + kb + (half << 4));
        af.q[1] = *(const uint4*)(arow + kb + 8 + (half << 4));

        // B fragments: col N = lane%16, K = half*16 + e (32B contiguous)
        const int kcol = kb + (half << 4);
        const v16bf b0 = *(const v16bf*)(ldsW + ( 0 + nlo) * KDIM + kcol);
        const v16bf b1 = *(const v16bf*)(ldsW + (16 + nlo) * KDIM + kcol);
        const v16bf b2 = *(const v16bf*)(ldsW + (32 + nlo) * KDIM + kcol);
        const v16bf b3 = *(const v16bf*)(ldsW + (48 + nlo) * KDIM + kcol);

        acc0 = __builtin_amdgcn_wmma_f32_16x16x32_bf16(false, af.v, false, b0, (short)0, acc0, false, false);
        acc1 = __builtin_amdgcn_wmma_f32_16x16x32_bf16(false, af.v, false, b1, (short)0, acc1, false, false);
        acc2 = __builtin_amdgcn_wmma_f32_16x16x32_bf16(false, af.v, false, b2, (short)0, acc2, false, false);
        acc3 = __builtin_amdgcn_wmma_f32_16x16x32_bf16(false, af.v, false, b3, (short)0, acc3, false, false);
    }

    // ---- epilogue: D layout -> N = lane%16, M = r + 8*half ----
    const float bb0 = biasc[Nbase +  0 + nlo];
    const float bb1 = biasc[Nbase + 16 + nlo];
    const float bb2 = biasc[Nbase + 32 + nlo];
    const float bb3 = biasc[Nbase + 48 + nlo];
#pragma unroll
    for (int r = 0; r < 8; ++r) {
        const int m  = blockIdx.x * 128 + wave * 16 + r + (half << 3);
        const int b  = m / HWS;
        const int hw = m - b * HWS;
        float* q = qkv + (size_t)b * QKV_B_STRIDE + hw;
        q[(size_t)(Nbase +  0 + nlo) * HWS] = acc0[r] + bb0;
        q[(size_t)(Nbase + 16 + nlo) * HWS] = acc1[r] + bb1;
        q[(size_t)(Nbase + 32 + nlo) * HWS] = acc2[r] + bb2;
        q[(size_t)(Nbase + 48 + nlo) * HWS] = acc3[r] + bb3;
    }
}

// ---------------- 4) BatchNorm stats -> folded scale/shift -----------------
__global__ __launch_bounds__(256) void bn_stats(
    const float* __restrict__ qkv,
    const float* __restrict__ gQ, const float* __restrict__ betaQ,
    const float* __restrict__ gK, const float* __restrict__ betaK,
    const float* __restrict__ gV, const float* __restrict__ betaV,
    float* __restrict__ scale, float* __restrict__ shift) {
    const int c = blockIdx.x;                 // channel 0..767
    const int tid = threadIdx.x;
    float s = 0.f, s2 = 0.f;
    for (int b = 0; b < BATCH; ++b) {
        const float* p = qkv + (size_t)b * QKV_B_STRIDE + (size_t)c * HWS;
        for (int hw = tid; hw < HWS; hw += 256) {
            float v = p[hw];
            s += v; s2 += v * v;
        }
    }
#pragma unroll
    for (int o = 16; o > 0; o >>= 1) {
        s  += __shfl_xor(s,  o, 32);
        s2 += __shfl_xor(s2, o, 32);
    }
    __shared__ float ls[8], ls2[8];
    if ((tid & 31) == 0) { ls[tid >> 5] = s; ls2[tid >> 5] = s2; }
    __syncthreads();
    if (tid == 0) {
        float S = 0.f, S2 = 0.f;
#pragma unroll
        for (int i = 0; i < 8; ++i) { S += ls[i]; S2 += ls2[i]; }
        const float inv = 1.0f / (float)M_TOT;
        const float mu  = S * inv;
        const float var = fmaxf(S2 * inv - mu * mu, 0.f);   // biased var (ddof=0)
        float g, be;
        if (c < 256)      { g = gQ[c];       be = betaQ[c]; }
        else if (c < 512) { g = gK[c - 256]; be = betaK[c - 256]; }
        else              { g = gV[c - 512]; be = betaV[c - 512]; }
        const float sc = g * rsqrtf(var + 1e-5f);
        scale[c] = sc;
        shift[c] = be - mu * sc;
    }
}

// ---------------- 5) 4-neighbor attention + ReLU + residual ----------------
// grid = 184832 blocks, 256 threads = 8 waves; wave <-> head, lane <-> channel.
__global__ __launch_bounds__(256) void attn(
    const float* __restrict__ qkv, const float* __restrict__ scale,
    const float* __restrict__ shift, const float* __restrict__ x,
    float* __restrict__ out) {
    const int pos = blockIdx.x;
    const int b  = pos / HWS;
    const int hw = pos - b * HWS;
    const int h  = hw / WWD;
    const int w  = hw - h * WWD;
    const int lane = threadIdx.x & 31;
    const int head = threadIdx.x >> 5;
    const int cq = head * ATT_C + lane;       // 0..255 (also out/residual channel)
    const int ck = 256 + cq;
    const int cv = 512 + cq;                  // OUT_C == ATT_C == 32

    const float* base = qkv + (size_t)b * QKV_B_STRIDE;
    const float qv = base[(size_t)cq * HWS + hw] * scale[cq] + shift[cq];
    const float sck = scale[ck], shk = shift[ck];
    const float scv = scale[cv], shv = shift[cv];

    const int dh[4] = {-1, 1, 0, 0};
    const int dw[4] = { 0, 0, -1, 1};
    float logit[4]; int nhw[4]; bool valid[4];
#pragma unroll
    for (int n = 0; n < 4; ++n) {
        const int h2 = h + dh[n], w2 = w + dw[n];
        const bool v = (h2 >= 0) & (h2 < HH) & (w2 >= 0) & (w2 < WWD);
        valid[n] = v;
        nhw[n] = h2 * WWD + w2;
        // zero-padding is applied AFTER BN -> OOB neighbor is literal 0
        float kk = v ? (base[(size_t)ck * HWS + nhw[n]] * sck + shk) : 0.f;
        float p = qv * kk;
#pragma unroll
        for (int o = 16; o > 0; o >>= 1) p += __shfl_xor(p, o, 32);
        logit[n] = v ? p * 0.17677669529663687f : 0.f;  // mean/32 * sqrt(32)
    }
    const float mx = fmaxf(fmaxf(logit[0], logit[1]), fmaxf(logit[2], logit[3]));
    float e[4], se = 0.f;
#pragma unroll
    for (int n = 0; n < 4; ++n) { e[n] = expf(logit[n] - mx); se += e[n]; }
    const float inv = 1.f / se;
    float o = 0.f;
#pragma unroll
    for (int n = 0; n < 4; ++n) {
        if (valid[n])
            o += e[n] * inv * (base[(size_t)cv * HWS + nhw[n]] * scv + shv);
    }
    o = fmaxf(o, 0.f);                                   // ReLU
    const size_t oi = (size_t)b * X_B_STRIDE + (size_t)cq * HWS + hw;
    out[oi] = o + x[oi];                                 // residual
}

// ---------------------------------------------------------------------------
extern "C" void kernel_launch(void* const* d_in, const int* in_sizes, int n_in,
                              void* d_out, int out_size, void* d_ws, size_t ws_size,
                              hipStream_t stream) {
    const float* x     = (const float*)d_in[0];
    const float* Wq    = (const float*)d_in[1];
    const float* bq    = (const float*)d_in[2];
    const float* Wk    = (const float*)d_in[3];
    const float* bk    = (const float*)d_in[4];
    const float* Wv    = (const float*)d_in[5];
    const float* bv    = (const float*)d_in[6];
    const float* gQ    = (const float*)d_in[7];
    const float* betaQ = (const float*)d_in[8];
    const float* gK    = (const float*)d_in[9];
    const float* betaK = (const float*)d_in[10];
    const float* gV    = (const float*)d_in[11];
    const float* betaV = (const float*)d_in[12];
    float* out = (float*)d_out;

    // Workspace layout (~663 MB total)
    char* ws = (char*)d_ws;
    size_t off = 0;
    unsigned short* xb = (unsigned short*)(ws + off);
    off += (size_t)BATCH * IN_C * HWS * sizeof(unsigned short);   // 94.6 MB
    off = (off + 255) & ~(size_t)255;
    unsigned short* wc = (unsigned short*)(ws + off);
    off += (size_t)NQKV * KDIM * sizeof(unsigned short);          // 384 KB
    off = (off + 255) & ~(size_t)255;
    float* biasc = (float*)(ws + off); off += NQKV * sizeof(float);
    off = (off + 255) & ~(size_t)255;
    float* scale = (float*)(ws + off); off += NQKV * sizeof(float);
    off = (off + 255) & ~(size_t)255;
    float* shift = (float*)(ws + off); off += NQKV * sizeof(float);
    off = (off + 255) & ~(size_t)255;
    float* qkv = (float*)(ws + off);                              // 567.8 MB
    off += (size_t)BATCH * NQKV * HWS * sizeof(float);
    (void)off; (void)ws_size; (void)in_sizes; (void)n_in; (void)out_size;

    const int n4 = (BATCH * IN_C * HWS) / 4;                      // 11,829,248
    convert_x<<<(n4 + 255) / 256, 256, 0, stream>>>(x, xb, n4);

    pack_weights<<<(NQKV * KDIM + 255) / 256, 256, 0, stream>>>(
        Wq, Wk, Wv, bq, bk, bv, wc, biasc);

    dim3 ggrid(M_TOT / 128, NQKV / 64);                           // (1444, 12)
    gemm_qkv<<<ggrid, 256, 0, stream>>>(xb, wc, biasc, qkv);

    bn_stats<<<NQKV, 256, 0, stream>>>(qkv, gQ, betaQ, gK, betaK, gV, betaV,
                                       scale, shift);

    attn<<<M_TOT, 256, 0, stream>>>(qkv, scale, shift, x, out);
}
